// LocalGOCorrOpt_40209483825171
// MI455X (gfx1250) — matrix operand order repaired and checked by reading, not compiled, not run on hardware
//
#include <hip/hip_runtime.h>

// ---------------------------------------------------------------------------
// LocalGOCorrOpt on MI455X (gfx1250, wave32, WMMA + Tensor Data Mover).
//
// Local correlation over c=128 -> v_wmma_f32_16x16x32_bf16 band-GEMM:
//   S[m,n] = sum_c f[c, m0+m] * ref[c, n0+n],  n0 = m0-8 (16x32 window),
// scores(dx) = diagonals of S; adjoint is a clean GEMM with a 32x16 band
// matrix T built from `mapped`. One wave32 per (batch, row, 16-px tile).
//
// Data movement: ref pre-padded into two bf16 global layouts; f/g bf16 copies
// stored channel-last. Every per-dy tile is an in-bounds 2-D rectangle moved
// by TENSOR_LOAD_TO_LDS (double-buffered, S_WAIT_TENSORCNT), and every WMMA
// operand (A and B) is read from LDS as two ds_load_b128.
// ---------------------------------------------------------------------------

typedef __attribute__((ext_vector_type(16))) __bf16        v16bf;
typedef __attribute__((ext_vector_type(8)))  float         v8f;
typedef __attribute__((ext_vector_type(4)))  unsigned int  v4u;
typedef __attribute__((ext_vector_type(8)))  int           v8i;
typedef __attribute__((ext_vector_type(4)))  int           v4i;

#define B_   4
#define C_   128
#define H_   128
#define W_   128
#define HW_  (H_*W_)
#define ND_  81
#define HP_  136      // H + 8 (4 rows of zero pad top/bottom)
#define WP_  160      // >= W + 8 + 24, 8B-friendly row pitch

__device__ __forceinline__ unsigned short f2bf_raw(float x) {
    unsigned u = __builtin_bit_cast(unsigned, x);
    unsigned r = u + 0x7FFFu + ((u >> 16) & 1u);   // round-to-nearest-even
    return (unsigned short)(r >> 16);
}

// Two 16B LDS reads -> one v16bf WMMA operand.
struct __attribute__((aligned(16))) bfpack { uint4 a, b; };
__device__ __forceinline__ v16bf ld16(const char* p0, const char* p1) {
    bfpack t;
    t.a = *(const uint4*)p0;
    t.b = *(const uint4*)p1;
    return __builtin_bit_cast(v16bf, t);
}

// ------------------------- Tensor Data Mover helpers -----------------------
#if __has_builtin(__builtin_amdgcn_tensor_load_to_lds) && \
    __has_builtin(__builtin_amdgcn_s_wait_tensorcnt)
#define TDM_AVAILABLE 1
#else
#define TDM_AVAILABLE 0
#endif

// 2-D tile DMA: y rows of (x8 * 8) bytes, global row stride (stride8 * 8) B.
__device__ __forceinline__ void tile_load(char* smem, unsigned lds_off,
                                          const void* gsrc, unsigned x8,
                                          unsigned y, unsigned stride8, int lane) {
#if TDM_AVAILABLE
    unsigned long long ga = (unsigned long long)(size_t)gsrc;
    v4u g0;
    g0[0] = 1u;                                            // count = 1 descriptor
    g0[1] = lds_off;                                       // lds_addr (bytes)
    g0[2] = (unsigned)ga;                                  // global_addr[31:0]
    g0[3] = (unsigned)((ga >> 32) & 0x1FFFFFFull) | (2u << 30);  // [56:32] | type=2
    v8i g1;
    g1[0] = (int)(3u << 16);          // workgroup_mask=0, data_size=3 (8 bytes)
    g1[1] = (int)(x8 << 16);          // tensor_dim0[15:0]  (exact fit, pre-padded)
    g1[2] = (int)(y  << 16);          // tensor_dim0[31:16]=0 | tensor_dim1[15:0]
    g1[3] = (int)(x8 << 16);          // tensor_dim1[31:16]=0 | tile_dim0
    g1[4] = (int)y;                   // tile_dim1 | tile_dim2=0
    g1[5] = (int)stride8;             // tensor_dim0_stride[31:0] (8B units)
    g1[6] = 0;                        // stride0[47:32] | dim1_stride[15:0]
    g1[7] = 0;                        // dim1_stride[47:16]
    v4i z4 = {0, 0, 0, 0};
#if defined(__clang_major__) && (__clang_major__ >= 23)
    v8i z8 = {0, 0, 0, 0, 0, 0, 0, 0};
    __builtin_amdgcn_tensor_load_to_lds(g0, g1, z4, z4, z8, 0);
#else
    __builtin_amdgcn_tensor_load_to_lds(g0, g1, z4, z4, 0);
#endif
#else
    // Fallback: cooperative copy through VGPRs.
    for (unsigned row = (unsigned)lane; row < y; row += 32) {
        const char* s = (const char*)gsrc + (size_t)row * stride8 * 8u;
        char* d = smem + lds_off + (size_t)row * x8 * 8u;
        for (unsigned q = 0; q < x8; ++q)
            ((unsigned long long*)d)[q] = ((const unsigned long long*)s)[q];
    }
#endif
}
__device__ __forceinline__ void tile_wait_le1() {
#if TDM_AVAILABLE
    __builtin_amdgcn_s_wait_tensorcnt(1);
#else
    __syncthreads();
#endif
}
__device__ __forceinline__ void tile_wait_le0() {
#if TDM_AVAILABLE
    __builtin_amdgcn_s_wait_tensorcnt(0);
#else
    __syncthreads();
#endif
}

// ----------------------- per-displacement constants ------------------------
__global__ void k_setup(const float* __restrict__ lw, const float* __restrict__ sw,
                        const float* __restrict__ mw, float4* __restrict__ consts) {
    int d = threadIdx.x;
    if (d >= ND_) return;
    int y = d / 9, x = d % 9;
    float dyf = (float)(y - 4), dxf = (float)(x - 4);
    float dist = sqrtf(dyf * dyf + dxf * dxf) * 2.0f;  // / BIN_DISP(0.5)
    float t = 0.f, vp = 0.f, mm = 0.f;
#pragma unroll
    for (int k = 0; k < 10; ++k) {
        float wv = (k == 9) ? fminf(fmaxf(dist - 8.0f, 0.f), 1.f)
                            : fmaxf(1.0f - fabsf(dist - (float)k), 0.f);
        t  += wv * lw[k];
        vp += wv * sw[k];
        mm += wv * mw[k];
    }
    float wmv = 1.0f / (1.0f + expf(-mm));
    consts[d] = make_float4(t, vp, 0.5f * (1.0f - wmv), 0.5f * (1.0f + wmv));
}

// ------------- init: f copy + channel-last bf16 cast of f ------------------
__global__ void k_init(const float* __restrict__ f_in, float* __restrict__ f_out,
                       unsigned short* __restrict__ fBT, int n) {
    int i = blockIdx.x * blockDim.x + threadIdx.x;
    if (i >= n) return;
    float fv = f_in[i];
    f_out[i] = fv;
    int wq = i % W_;
    int hq = (i / W_) % H_;
    int cq = (i / HW_) % C_;
    int bq = i / (C_ * HW_);
    fBT[(((size_t)bq * H_ + hq) * W_ + wq) * C_ + cq] = f2bf_raw(fv);
}

// ------------- build zero-padded bf16 ref in both layouts ------------------
// refP [b][c][hp][wp] (channel-major)  and  refT [b][hp][wp][c] (channel-last)
__global__ void k_pad(const float* __restrict__ ref, unsigned short* __restrict__ refP,
                      unsigned short* __restrict__ refT) {
    int i = blockIdx.x * blockDim.x + threadIdx.x;
    const int NP = B_ * C_ * HP_ * WP_;
    if (i >= NP) return;
    int wp = i % WP_;
    int hp = (i / WP_) % HP_;
    int c  = (i / (WP_ * HP_)) % C_;
    int bb = i / (WP_ * HP_ * C_);
    int h = hp - 4, w = wp - 8;
    unsigned short v = 0;
    if (h >= 0 && h < H_ && w >= 0 && w < W_)
        v = f2bf_raw(ref[(((size_t)bb * C_ + c) * H_ + h) * W_ + w]);
    refP[i] = v;
    refT[(((size_t)bb * HP_ + hp) * WP_ + wp) * C_ + c] = v;
}

// ----------------------------- corr (WMMA) ---------------------------------
__global__ __launch_bounds__(32)
void k_corr(const unsigned short* __restrict__ xBT,   // [b][h][w][c] bf16
            const unsigned short* __restrict__ refT,  // [b][hp][wp][c] bf16
            float* __restrict__ scores) {
    extern __shared__ char smem[];
    // [0,8192) ref tile buf0 [wp2=32][c=128]; [8192,16384) buf1;
    // [16384,20480) fA [px=16][c=128]; [20480,22528) S [16][32] f32
    const unsigned BUF0 = 0u, BUF1 = 8192u, FA = 16384u, SOFF = 20480u;

    int lane = threadIdx.x;
    int bid  = blockIdx.x;
    int m0   = (bid & 7) * 16;
    int h    = (bid >> 3) & (H_ - 1);
    int bb   = bid >> 10;
    int m    = lane & 15, half = lane >> 4;

    // TDM: f tile (A operand) + first ref tile, in flight together.
    const unsigned short* fsrc = xBT + (((size_t)bb * H_ + h) * W_ + m0) * C_;
    tile_load(smem, FA, fsrc, 32, 16, 32, lane);            // 16 rows x 256B
    const unsigned short* r0 = refT + ((((size_t)bb * HP_ + (h + 0)) * WP_ + m0) * C_);
    tile_load(smem, BUF0, r0, 32, 32, 32, lane);            // 32 rows x 256B
    tile_wait_le1();                                        // fA ready

    // A chunks (M=px, K=ch): one contiguous 16-element run each -> 2x b128.
    v16bf a[4];
#pragma unroll
    for (int kc = 0; kc < 4; ++kc) {
        const char* ap = smem + FA + 2 * (m * 128 + kc * 32 + 16 * half);
        a[kc] = ld16(ap, ap + 16);
    }

    float* S = (float*)(smem + SOFF);
    for (int dy = 0; dy < 9; ++dy) {
        if (dy < 8) {       // prefetch next dy, then wait for current
            const unsigned short* rn =
                refT + ((((size_t)bb * HP_ + (h + dy + 1)) * WP_ + m0) * C_);
            tile_load(smem, ((dy + 1) & 1) ? BUF1 : BUF0, rn, 32, 32, 32, lane);
            tile_wait_le1();
        } else {
            tile_wait_le0();
        }
        const char* buf = smem + ((dy & 1) ? BUF1 : BUF0);

        v8f c0 = {}; v8f c1 = {};
#pragma unroll
        for (int kc = 0; kc < 4; ++kc) {
            int ko = kc * 32 + 16 * half;                     // K base for this lane
            const char* p0 = buf + 2 * (m * 128 + ko);        // B tile0: n = m
            const char* p1 = buf + 2 * ((16 + m) * 128 + ko); // B tile1: n = 16+m
            v16bf b0 = ld16(p0, p0 + 16);
            v16bf b1 = ld16(p1, p1 + 16);
            c0 = __builtin_amdgcn_wmma_f32_16x16x32_bf16(false, a[kc], false, b0,
                                                         (short)0, c0, false, false);
            c1 = __builtin_amdgcn_wmma_f32_16x16x32_bf16(false, a[kc], false, b1,
                                                         (short)0, c1, false, false);
        }

        __syncthreads();   // previous extract reads of S are done
#pragma unroll
        for (int r = 0; r < 8; ++r) {     // C/D: M = r + 8*half, N = lane%16
            S[(r + 8 * half) * 32 + m]      = c0[r];
            S[(r + 8 * half) * 32 + 16 + m] = c1[r];
        }
        __syncthreads();

        // score(m2, dx) = S[m2][m2 + dx + 4]; 144 = 9 dx * 16 px
        for (int idx = lane; idx < 144; idx += 32) {
            int dx = idx >> 4, m2 = idx & 15;
            float v = S[m2 * 32 + (m2 + dx + 4)];
            scores[(((size_t)bb * ND_ + dy * 9 + dx) * H_ + h) * W_ + m0 + m2] = v;
        }
    }
}

// ------------------- elementwise: scores -> grad_act, mapped ---------------
__global__ void k_map(const float* __restrict__ scores, const float4* __restrict__ consts,
                      float* __restrict__ gact, unsigned short* __restrict__ mapped, int n) {
    int i = blockIdx.x * blockDim.x + threadIdx.x;
    if (i >= n) return;
    float s = scores[i];
    int d = (i >> 14) % ND_;                 // HW = 16384
    float4 c = consts[d];
    float sgn   = (s > 0.f) ? 1.f : ((s < 0.f) ? -1.f : 0.f);
    float act   = c.y * (c.z * fabsf(s) + c.w * s);
    float ga    = c.y * (c.z * sgn + c.w);
    float resid = act - c.y * c.x;
    gact[i]   = ga;
    mapped[i] = f2bf_raw(ga * resid);
}

// --------------------------- corr_t (WMMA) ---------------------------------
__global__ __launch_bounds__(32)
void k_corrt(const unsigned short* __restrict__ mapped,
             const unsigned short* __restrict__ refP,
             const float* __restrict__ f_cur, const float* __restrict__ filter_reg,
             float* __restrict__ fgrad, unsigned short* __restrict__ gBT) {
    extern __shared__ char smem[];
    // [0,8192) ref tile buf0 [c=128][wp2=32]; [8192,16384) buf1;
    // [16384,17408) T [m=16][n=32] bf16 (zero-padded band matrix)
    const unsigned BUF0 = 0u, BUF1 = 8192u, TOFF = 16384u;

    int lane = threadIdx.x;
    int bid  = blockIdx.x;
    int m0   = (bid & 7) * 16;
    int h    = (bid >> 3) & (H_ - 1);
    int bb   = bid >> 10;
    int m    = lane & 15, half = lane >> 4;

    const unsigned short* r0 =
        refP + (((size_t)bb * C_ * HP_ + (h + 0)) * WP_ + m0);
    tile_load(smem, BUF0, r0, 8, 128, 5440, lane);           // 128 rows x 64B

    v8f acc[8];
    v8f zero = {};
#pragma unroll
    for (int mc = 0; mc < 8; ++mc) acc[mc] = zero;

    char* T = smem + TOFF;
    for (int dy = 0; dy < 9; ++dy) {
        if (dy < 8) {
            const unsigned short* rn =
                refP + (((size_t)bb * C_ * HP_ + (h + dy + 1)) * WP_ + m0);
            tile_load(smem, ((dy + 1) & 1) ? BUF1 : BUF0, rn, 8, 128, 5440, lane);
            tile_wait_le1();
        } else {
            tile_wait_le0();
        }
        const char* buf = smem + ((dy & 1) ? BUF1 : BUF0);

        // Build band matrix T[m][n] (1 KB): zero-fill, then scatter mapped.
        __syncthreads();                        // prior B reads of T done
        {
            uint4 z4 = make_uint4(0u, 0u, 0u, 0u);
            uint4* tz = (uint4*)(T + lane * 32);
            tz[0] = z4; tz[1] = z4;
            for (int idx = lane; idx < 144; idx += 32) {
                int dx = idx >> 4, m2 = idx & 15;
                unsigned short v =
                    mapped[(((size_t)bb * ND_ + dy * 9 + dx) * H_ + h) * W_ + m0 + m2];
                *(unsigned short*)(T + 2 * (m2 * 32 + (m2 + dx + 4))) = v;
            }
        }
        __syncthreads();

        // B operand: T rows, K = n = e + 16*half (contiguous 16 elements).
        const char* tp = T + 2 * (m * 32 + 16 * half);
        v16bf bq = ld16(tp, tp + 16);
#pragma unroll
        for (int mc = 0; mc < 8; ++mc) {
            // A operand: M = channel (crow), K = n; contiguous 16 elements.
            const char* ap = buf + 2 * ((mc * 16 + m) * 32 + 16 * half);
            v16bf av = ld16(ap, ap + 16);
            acc[mc] = __builtin_amdgcn_wmma_f32_16x16x32_bf16(false, av, false, bq,
                                                              (short)0, acc[mc], false, false);
        }
    }

    float fr  = filter_reg[0];
    float reg = fmaxf(fr * fr, 1e-10f) * (1.0f / ((float)C_ * (float)C_));
    size_t tbase = (((size_t)bb * H_ + h) * W_ + m0 + m) * C_;
#pragma unroll
    for (int mc = 0; mc < 8; ++mc) {
#pragma unroll
        for (int r = 0; r < 8; ++r) {
            int ch = mc * 16 + r + 8 * half;
            size_t idx = (((size_t)(bb * C_ + ch)) * H_ + h) * W_ + m0 + m;
            float g = reg * f_cur[idx] + acc[mc][r];
            fgrad[idx] = g;
            gBT[tbase + ch] = f2bf_raw(g);
        }
    }
}

// ---------------- alpha denominator: sum_d (grad_act*corr)^2 ----------------
__global__ void k_sg(const float* __restrict__ scores, const float* __restrict__ gact,
                     float* __restrict__ aden) {
    int i = blockIdx.x * blockDim.x + threadIdx.x;
    if (i >= B_ * HW_) return;
    int bb = i >> 14, p = i & (HW_ - 1);
    size_t base = (size_t)bb * ND_ * HW_ + p;
    float acc = 0.f;
    for (int d = 0; d < ND_; ++d) {
        size_t idx = base + (size_t)d * HW_;
        float t = scores[idx] * gact[idx];
        acc += t * t;
    }
    aden[i] = acc;
}

// --------------------------- filter update ---------------------------------
__global__ void k_update(const float* __restrict__ fgrad, const float* __restrict__ aden,
                         const float* __restrict__ filter_reg, const float* __restrict__ log_step,
                         float* __restrict__ f_out, unsigned short* __restrict__ fBT) {
    int i = blockIdx.x * blockDim.x + threadIdx.x;
    if (i >= B_ * HW_) return;
    int bb = i >> 14, p = i & (HW_ - 1);
    size_t base = (size_t)bb * C_ * HW_ + p;
    float num = 0.f;
    for (int c = 0; c < C_; ++c) {
        float g = fgrad[base + (size_t)c * HW_];
        num += g * g;
    }
    float fr  = filter_reg[0];
    float reg = fmaxf(fr * fr, 1e-10f) * (1.0f / ((float)C_ * (float)C_));
    float den = aden[i] + reg * num;
    float alpha = num / fmaxf(den, 1e-8f);
    float sa = expf(log_step[0]) * alpha;
    size_t tbase = ((size_t)bb * HW_ + p) * C_;   // channel-last bf16 copy
    for (int c = 0; c < C_; ++c) {
        size_t idx = base + (size_t)c * HW_;
        float fn = f_out[idx] - sa * fgrad[idx];
        f_out[idx] = fn;
        fBT[tbase + c] = f2bf_raw(fn);
    }
}

// ---------------------------------------------------------------------------
extern "C" void kernel_launch(void* const* d_in, const int* in_sizes, int n_in,
                              void* d_out, int out_size, void* d_ws, size_t ws_size,
                              hipStream_t stream) {
    const float* filter_map = (const float*)d_in[0];
    const float* ref_feat   = (const float*)d_in[1];
    const float* label_w    = (const float*)d_in[2];
    const float* spatial_w  = (const float*)d_in[3];
    const float* mask_w     = (const float*)d_in[4];
    const float* log_step   = (const float*)d_in[5];
    const float* filter_reg = (const float*)d_in[6];
    float* f_out = (float*)d_out;

    const size_t N   = (size_t)B_ * C_ * H_ * W_;        // 8,388,608
    const size_t S81 = (size_t)B_ * ND_ * H_ * W_;       // 5,308,416
    const size_t NP  = (size_t)B_ * C_ * HP_ * WP_;      // 11,141,120

    char* w = (char*)d_ws;
    auto take = [&](size_t bytes) {
        char* p = w;
        w += (bytes + 255) & ~(size_t)255;
        return p;
    };
    unsigned short* refP   = (unsigned short*)take(NP * 2);   // [b][c][hp][wp]
    unsigned short* refT   = (unsigned short*)take(NP * 2);   // [b][hp][wp][c]
    unsigned short* fBT    = (unsigned short*)take(N * 2);    // [b][h][w][c]
    unsigned short* gBT    = (unsigned short*)take(N * 2);    // [b][h][w][c]
    float*          fgrad  = (float*)take(N * 4);
    float*          scores = (float*)take(S81 * 4);
    float*          gact   = (float*)take(S81 * 4);
    unsigned short* mapped = (unsigned short*)take(S81 * 2);
    float*          aden   = (float*)take((size_t)B_ * HW_ * 4);
    float4*         consts = (float4*)take(ND_ * sizeof(float4));

    hipLaunchKernelGGL(k_setup, dim3(1), dim3(96), 0, stream,
                       label_w, spatial_w, mask_w, consts);
    hipLaunchKernelGGL(k_init, dim3((unsigned)(N / 256)), dim3(256), 0, stream,
                       filter_map, f_out, fBT, (int)N);
    hipLaunchKernelGGL(k_pad, dim3((unsigned)(NP / 256)), dim3(256), 0, stream,
                       ref_feat, refP, refT);

    const unsigned corrGrid = B_ * H_ * 8;   // 4096 single-wave workgroups
    for (int it = 0; it < 3; ++it) {
        hipLaunchKernelGGL(k_corr, dim3(corrGrid), dim3(32), 22528, stream,
                           fBT, refT, scores);
        hipLaunchKernelGGL(k_map, dim3((unsigned)(S81 / 256)), dim3(256), 0, stream,
                           scores, consts, gact, mapped, (int)S81);
        hipLaunchKernelGGL(k_corrt, dim3(corrGrid), dim3(32), 17408, stream,
                           mapped, refP, f_out, filter_reg, fgrad, gBT);
        hipLaunchKernelGGL(k_corr, dim3(corrGrid), dim3(32), 22528, stream,
                           gBT, refT, scores);
        hipLaunchKernelGGL(k_sg, dim3((B_ * HW_) / 256), dim3(256), 0, stream,
                           scores, gact, aden);
        hipLaunchKernelGGL(k_update, dim3((B_ * HW_) / 256), dim3(256), 0, stream,
                           fgrad, aden, filter_reg, log_step, f_out, fBT);
    }
}
